// STU_43739946943379
// MI455X (gfx1250) — matrix operand order repaired and checked by reading, not compiled, not run on hardware
//
#include <hip/hip_runtime.h>
#include <math.h>

// ---------------------------------------------------------------------------
// STU layer for MI455X (gfx1250, wave32, WMMA f16 -> f32 accum).
//   delta_phi : Toeplitz-as-GEMM spectral conv  (A-fragments pre-swizzled)
//   delta_ar  : 3-tap AR GEMM (fused with delta combine)
//   scan      : log-depth doubling of transfer matrices Psi_s (S=64 window,
//               ||Psi_s|| ~ 0.75^s so truncation error ~1e-8), then one
//               windowed WMMA GEMM  y[t] = sum_s Psi_s @ delta[t-s].
// v3: revert manual B double-buffering (caused accumulator mov churn /
//     fewer unrolled WMMAs in v2); keep LDS-tiled transpose + fused epilogue.
// ---------------------------------------------------------------------------

typedef _Float16 hv8  __attribute__((ext_vector_type(8)));
typedef _Float16 hv16 __attribute__((ext_vector_type(16)));
typedef float    fv8  __attribute__((ext_vector_type(8)));

#define DEVI __device__ __forceinline__

constexpr int L_  = 2048;   // sequence length
constexpr int D_  = 768;    // model dim
constexpr int KF_ = 24;     // spectral filters
constexpr int S_  = 64;     // scan truncation window

// ---------------- workspace layout (bytes, all 256-aligned) ----------------
constexpr size_t SZ_UBF   = (size_t)L_ * D_ * 2;            // inputs as f16
constexpr size_t SZ_MPHIT = (size_t)KF_ * D_ * D_ * 2;      // m_phi^T per k  [k][o][d]
constexpr size_t SZ_AF    = (size_t)KF_ * 128 * 32 * 16 * 2;// Toeplitz A-frags
constexpr size_t SZ_BT    = (size_t)KF_ * D_ * L_ * 2;      // Bt[k][o][t']
constexpr size_t SZ_MUT   = (size_t)3 * D_ * D_ * 2;        // m_u^T  [kk][o][d]
constexpr size_t SZ_MYT   = (size_t)2 * D_ * D_ * 2;        // m_y    [i][m][d]
constexpr size_t SZ_DPHI  = (size_t)L_ * D_ * 4;            // f32
constexpr size_t SZ_DBF   = (size_t)L_ * D_ * 2;            // delta as f16
constexpr size_t SZ_Q     = (size_t)D_ * D_ * 2;            // doubling temp
constexpr size_t SZ_PSI   = (size_t)S_ * D_ * D_ * 2;       // Psi_s  [s][o][d]

constexpr size_t OFF_UBF   = 0;
constexpr size_t OFF_MPHIT = OFF_UBF   + SZ_UBF;
constexpr size_t OFF_AF    = OFF_MPHIT + SZ_MPHIT;
constexpr size_t OFF_BT    = OFF_AF    + SZ_AF;
constexpr size_t OFF_MUT   = OFF_BT    + SZ_BT;
constexpr size_t OFF_MYT   = OFF_MUT   + SZ_MUT;
constexpr size_t OFF_DPHI  = OFF_MYT   + SZ_MYT;
constexpr size_t OFF_DBF   = OFF_DPHI  + SZ_DPHI;
constexpr size_t OFF_Q     = OFF_DBF   + SZ_DBF;
constexpr size_t OFF_PSI   = OFF_Q     + SZ_Q;
constexpr size_t OFF_END   = OFF_PSI   + SZ_PSI;            // ~202 MB

// ---------------- WMMA fragment helpers (wave32, contiguous k order shared by A and B) -------
// A (16x32 f16): lane m = lane&15, hi = lane>>4; element e -> K = e+16*hi
DEVI hv16 h16z() { hv16 z = {}; return z; }

DEVI hv16 lda_row(const _Float16* __restrict__ A, int lda, int row, int kb, int hi) {
    const _Float16* p = A + (size_t)row * lda + kb + 16 * hi;
    return *(const hv16*)p;
}

// B (32x16 f16), stored N-major [n][k]: lane col n = lane&15; element e -> K = kb+16*hi+e
DEVI hv16 ldb_nmaj(const _Float16* __restrict__ Bn, int ldb, int col, int kb, int hi) {
    return *(const hv16*)(Bn + (size_t)col * ldb + kb + 16 * hi);
}

// B stored K-major [k][n] (natural): strided gather (lanes stay coalesced in n)
DEVI hv16 ldb_str(const _Float16* __restrict__ B, int ldb, int kb, int col, int hi) {
    const _Float16* p = B + (size_t)(kb + 16 * hi) * ldb + col;
    hv16 b;
#pragma unroll
    for (int e = 0; e < 16; ++e) b[e] = p[(size_t)e * ldb];
    return b;
}

DEVI fv8 wmma16(hv16 a, hv16 b, fv8 c) {
    return __builtin_amdgcn_wmma_f32_16x16x32_f16(false, a, false, b, (short)0, c, false, false);
}

// ---------------- prep kernels ---------------------------------------------
__global__ void stu_cvt_u(const float* __restrict__ x, _Float16* __restrict__ o) {
    int i = blockIdx.x * 256 + threadIdx.x;
    if (i < L_ * D_) o[i] = (_Float16)x[i];
}
// LDS 32x32 tile transpose: mphiT[k][o][d] = m_phi[k*D+d][o], coalesced both ways
__global__ void stu_mk_mphiT(const float* __restrict__ mphi, _Float16* __restrict__ o) {
    __shared__ float tile[32][33];
    int k = blockIdx.y;
    int tx = blockIdx.x % (D_ / 32);        // o tile
    int ty = blockIdx.x / (D_ / 32);        // d tile
    int o0 = tx * 32, d0 = ty * 32;
    int jj = threadIdx.x & 31, ii = threadIdx.x >> 5;
#pragma unroll
    for (int r = 0; r < 4; ++r) {
        int d = d0 + ii + r * 8;
        tile[ii + r * 8][jj] = mphi[((size_t)(k * D_ + d)) * D_ + o0 + jj];
    }
    __syncthreads();
#pragma unroll
    for (int r = 0; r < 4; ++r) {
        int oo = o0 + ii + r * 8;
        o[((size_t)k * D_ + oo) * D_ + d0 + jj] = (_Float16)tile[jj][ii + r * 8];
    }
}
__global__ void stu_mk_mut(const float* __restrict__ mu, _Float16* __restrict__ o) {
    int i = blockIdx.x * 256 + threadIdx.x;
    if (i >= 3 * D_ * D_) return;
    int kk = i / (D_ * D_), r = i % (D_ * D_), oo = r / D_, d = r % D_;
    o[i] = (_Float16)mu[(size_t)oo * D_ * 3 + (size_t)d * 3 + kk];
}
__global__ void stu_mk_myt(const float* __restrict__ my, _Float16* __restrict__ o) {
    int i = blockIdx.x * 256 + threadIdx.x;
    if (i >= 2 * D_ * D_) return;
    int ii = i / (D_ * D_), r = i % (D_ * D_), mm = r / D_, d = r % D_;
    o[i] = (_Float16)my[(size_t)mm * (2 * D_) + (size_t)ii * D_ + d];
}
__global__ void stu_psi0(const float* __restrict__ my, _Float16* __restrict__ Psi) {
    int i = blockIdx.x * 256 + threadIdx.x;
    if (i >= D_ * D_) return;
    int oo = i / D_, d = i % D_;
    Psi[i] = (_Float16)((oo == d) ? 1.0f : 0.0f);                 // Psi_0 = I
    Psi[(size_t)D_ * D_ + i] = (_Float16)my[(size_t)oo * (2 * D_) + d]; // Psi_1 = M0
}
// Pre-swizzled Toeplitz A fragments: Af[k][g][lane][e], g = (16*t_tile-32*t'c)/16
__global__ void stu_afrag(const float* __restrict__ evecs, const float* __restrict__ evals,
                          _Float16* __restrict__ Af) {
    int gid = blockIdx.x * 256 + threadIdx.x;
    if (gid >= KF_ * 128 * 32) return;
    int k = gid / (128 * 32), rem = gid % (128 * 32), g = rem / 32, lane = rem % 32;
    int m = lane & 15, hi = lane >> 4;
    float sc = powf(evals[k], 0.25f);
    hv16 v;
#pragma unroll
    for (int e = 0; e < 16; ++e) {
        int kk = e + 16 * hi;
        int s = 16 * g + m - kk - 2;            // = (t - t') - 2
        float val = (s >= 0 && s < L_) ? evecs[(size_t)s * KF_ + k] * sc : 0.0f;
        v[e] = (_Float16)val;
    }
    *(hv16*)(Af + (size_t)gid * 16) = v;
}

// ---------------- stage 1: Bt[k][o][t'] = (U @ m_phi_k)^T -------------------
__global__ void stu_bt(const _Float16* __restrict__ ubf, const _Float16* __restrict__ mphiT,
                       _Float16* __restrict__ Bt) {
    int tid = threadIdx.x, lane = tid & 31, wv = tid >> 5;
    int wid = blockIdx.x * 8 + wv;
    const int perk = (L_ / 64) * (D_ / 16);
    int k = wid / perk, r = wid % perk, tsup = r / (D_ / 16), oT = r % (D_ / 16);
    int t0 = tsup * 64, o0 = oT * 16, ln = lane & 15, hi = lane >> 4;
    const _Float16* Bk = mphiT + (size_t)k * D_ * D_;
    fv8 acc[4] = {};
    for (int dc = 0; dc < D_ / 32; ++dc) {
        int kb = dc * 32;
        hv16 b = ldb_nmaj(Bk, D_, o0 + ln, kb, hi);
#pragma unroll
        for (int j = 0; j < 4; ++j) {
            hv16 a = lda_row(ubf, D_, t0 + j * 16 + ln, kb, hi);
            acc[j] = wmma16(a, b, acc[j]);
        }
    }
#pragma unroll
    for (int j = 0; j < 4; ++j) {
        hv8 hv;
#pragma unroll
        for (int v = 0; v < 8; ++v) hv[v] = (_Float16)acc[j][v];
        *(hv8*)(Bt + ((size_t)(k * D_ + o0 + ln)) * L_ + t0 + j * 16 + 8 * hi) = hv;
    }
}

// ---------------- stage 2: Toeplitz conv -> dphi ----------------------------
__global__ void stu_dphi(const _Float16* __restrict__ Af, const _Float16* __restrict__ Bt,
                         float* __restrict__ dphi) {
    int tid = threadIdx.x, lane = tid & 31, wv = tid >> 5;
    int wid = blockIdx.x * 8 + wv;
    int tsup = wid / (D_ / 16), oT = wid % (D_ / 16);
    int b16 = tsup * 4, t0 = tsup * 64, o0 = oT * 16, ln = lane & 15, hi = lane >> 4;
    int tcmax = (b16 + 3) >> 1;
    fv8 acc[4] = {};
    for (int k = 0; k < KF_; ++k) {
        const _Float16* BtK = Bt + (size_t)k * D_ * L_;
        const _Float16* AfK = Af + (size_t)k * 128 * 32 * 16;
        for (int tc = 0; tc <= tcmax; ++tc) {
            hv16 b = ldb_nmaj(BtK, L_, o0 + ln, tc * 32, hi);
#pragma unroll
            for (int j = 0; j < 4; ++j) {
                int tb16 = b16 + j;
                if (tc <= (tb16 >> 1)) {                     // wave-uniform branch
                    int g = tb16 - 2 * tc;
                    hv16 a = *(const hv16*)(AfK + ((size_t)g * 32 + lane) * 16);
                    acc[j] = wmma16(a, b, acc[j]);
                }
            }
        }
    }
#pragma unroll
    for (int j = 0; j < 4; ++j)
#pragma unroll
        for (int v = 0; v < 8; ++v)
            dphi[(size_t)(t0 + j * 16 + v + 8 * hi) * D_ + o0 + ln] = acc[j][v];
}

// ------- stage 3: AR term, fused with delta = dphi + dar -> delta_bf --------
__global__ void stu_ar(const _Float16* __restrict__ ubf, const _Float16* __restrict__ mut,
                       const float* __restrict__ dphi, _Float16* __restrict__ dbf) {
    int tid = threadIdx.x, lane = tid & 31, wv = tid >> 5;
    int wid = blockIdx.x * 8 + wv;
    int tsup = wid / (D_ / 16), oT = wid % (D_ / 16);
    int t0 = tsup * 64, o0 = oT * 16, ln = lane & 15, hi = lane >> 4;
    fv8 acc[4] = {};
    for (int kk = 0; kk < 3; ++kk) {
        const _Float16* Bk = mut + (size_t)kk * D_ * D_;
        for (int dc = 0; dc < D_ / 32; ++dc) {
            int kb = dc * 32;
            hv16 b = ldb_nmaj(Bk, D_, o0 + ln, kb, hi);
#pragma unroll
            for (int j = 0; j < 4; ++j) {
                int rrow = t0 + j * 16 + ln - kk;
                hv16 a = (rrow >= 0) ? lda_row(ubf, D_, rrow, kb, hi) : h16z();
                acc[j] = wmma16(a, b, acc[j]);
            }
        }
    }
#pragma unroll
    for (int j = 0; j < 4; ++j)
#pragma unroll
        for (int v = 0; v < 8; ++v) {
            size_t idx = (size_t)(t0 + j * 16 + v + 8 * hi) * D_ + o0 + ln;
            dbf[idx] = (_Float16)(acc[j][v] + dphi[idx]);
        }
}

// ---------------- scan: Psi doubling ---------------------------------------
// Q = Psi[n-2] @ M1   (per doubling stage)
__global__ void stu_q(const _Float16* __restrict__ Psi, const _Float16* __restrict__ myt,
                      _Float16* __restrict__ Q, int n) {
    int tid = threadIdx.x, lane = tid & 31, wv = tid >> 5;
    int wid = blockIdx.x * 8 + wv;
    int oT = wid / (D_ / 16), dT = wid % (D_ / 16);
    int o0 = oT * 16, d0 = dT * 16, ln = lane & 15, hi = lane >> 4;
    const _Float16* A = Psi + (size_t)(n - 2) * D_ * D_;
    const _Float16* B = myt + (size_t)D_ * D_;   // M1 natural [m][d]
    fv8 acc = {};
    for (int mc = 0; mc < D_ / 32; ++mc) {
        int kb = mc * 32;
        hv16 a = lda_row(A, D_, o0 + ln, kb, hi);
        hv16 b = ldb_str(B, D_, kb, d0 + ln, hi);
        acc = wmma16(a, b, acc);
    }
#pragma unroll
    for (int v = 0; v < 8; ++v)
        Q[(size_t)(o0 + v + 8 * hi) * D_ + d0 + ln] = (_Float16)acc[v];
}

// Psi[n-1+s] = Psi[n-1] @ Psi[s] + Q @ Psi[s-1],  s = blockIdx.y + 1
__global__ void stu_psi(_Float16* __restrict__ Psi, const _Float16* __restrict__ Q, int n) {
    int s = blockIdx.y + 1;
    int tid = threadIdx.x, lane = tid & 31, wv = tid >> 5;
    int wid = blockIdx.x * 8 + wv;
    int oT = wid / (D_ / 16), dT = wid % (D_ / 16);
    int o0 = oT * 16, d0 = dT * 16, ln = lane & 15, hi = lane >> 4;
    const _Float16* A1 = Psi + (size_t)(n - 1) * D_ * D_;
    const _Float16* B1 = Psi + (size_t)s * D_ * D_;
    const _Float16* B2 = Psi + (size_t)(s - 1) * D_ * D_;
    _Float16* out = Psi + (size_t)(n - 1 + s) * D_ * D_;
    fv8 acc = {};
    for (int mc = 0; mc < D_ / 32; ++mc) {
        int kb = mc * 32;
        hv16 a1 = lda_row(A1, D_, o0 + ln, kb, hi);
        hv16 b1 = ldb_str(B1, D_, kb, d0 + ln, hi);
        acc = wmma16(a1, b1, acc);
        hv16 a2 = lda_row(Q, D_, o0 + ln, kb, hi);
        hv16 b2 = ldb_str(B2, D_, kb, d0 + ln, hi);
        acc = wmma16(a2, b2, acc);
    }
#pragma unroll
    for (int v = 0; v < 8; ++v)
        out[(size_t)(o0 + v + 8 * hi) * D_ + d0 + ln] = (_Float16)acc[v];
}

// ---------------- final: y[t] = sum_{s<S} Psi_s @ delta[t-s] ----------------
__global__ void stu_final(const _Float16* __restrict__ dbf, const _Float16* __restrict__ Psi,
                          float* __restrict__ y) {
    int tid = threadIdx.x, lane = tid & 31, wv = tid >> 5;
    int wid = blockIdx.x * 8 + wv;
    int tsup = wid / (D_ / 16), oT = wid % (D_ / 16);
    int t0 = tsup * 64, o0 = oT * 16, ln = lane & 15, hi = lane >> 4;
    fv8 acc[4] = {};
    for (int s = 0; s < S_; ++s) {
        const _Float16* Ps = Psi + (size_t)s * D_ * D_;   // [o][d] = B^T layout
        for (int dc = 0; dc < D_ / 32; ++dc) {
            int kb = dc * 32;
            hv16 b = ldb_nmaj(Ps, D_, o0 + ln, kb, hi);
#pragma unroll
            for (int j = 0; j < 4; ++j) {
                int rrow = t0 + j * 16 + ln - s;
                hv16 a = (rrow >= 0) ? lda_row(dbf, D_, rrow, kb, hi) : h16z();
                acc[j] = wmma16(a, b, acc[j]);
            }
        }
    }
#pragma unroll
    for (int j = 0; j < 4; ++j)
#pragma unroll
        for (int v = 0; v < 8; ++v)
            y[(size_t)(t0 + j * 16 + v + 8 * hi) * D_ + o0 + ln] = acc[j][v];
}

// ---------------------------------------------------------------------------
extern "C" void kernel_launch(void* const* d_in, const int* in_sizes, int n_in,
                              void* d_out, int out_size, void* d_ws, size_t ws_size,
                              hipStream_t stream) {
    if (ws_size < OFF_END) return;   // need ~202 MB scratch
    const float* inputs  = (const float*)d_in[0];
    const float* m_y     = (const float*)d_in[1];
    const float* m_u     = (const float*)d_in[2];
    const float* m_phi   = (const float*)d_in[3];
    const float* e_vals  = (const float*)d_in[4];
    const float* e_vecs  = (const float*)d_in[5];
    float* y = (float*)d_out;

    char* w = (char*)d_ws;
    _Float16* ubf   = (_Float16*)(w + OFF_UBF);
    _Float16* mphiT = (_Float16*)(w + OFF_MPHIT);
    _Float16* Af    = (_Float16*)(w + OFF_AF);
    _Float16* Bt    = (_Float16*)(w + OFF_BT);
    _Float16* mut   = (_Float16*)(w + OFF_MUT);
    _Float16* myt   = (_Float16*)(w + OFF_MYT);
    float*    dphi  = (float*)(w + OFF_DPHI);
    _Float16* dbf   = (_Float16*)(w + OFF_DBF);
    _Float16* Q     = (_Float16*)(w + OFF_Q);
    _Float16* Psi   = (_Float16*)(w + OFF_PSI);

    // prep
    stu_cvt_u   <<<(L_ * D_ + 255) / 256, 256, 0, stream>>>(inputs, ubf);
    stu_mk_mphiT<<<dim3((D_ / 32) * (D_ / 32), KF_), 256, 0, stream>>>(m_phi, mphiT);
    stu_mk_mut  <<<(3 * D_ * D_ + 255) / 256, 256, 0, stream>>>(m_u, mut);
    stu_mk_myt  <<<(2 * D_ * D_ + 255) / 256, 256, 0, stream>>>(m_y, myt);
    stu_psi0    <<<(D_ * D_ + 255) / 256, 256, 0, stream>>>(m_y, Psi);
    stu_afrag   <<<(KF_ * 128 * 32 + 255) / 256, 256, 0, stream>>>(e_vecs, e_vals, Af);

    // Psi doubling chain: after stage (n,cnt) we hold Psi_0 .. Psi_{n-1+cnt}
    const int stages[6][2] = {{2,1},{3,2},{5,4},{9,8},{17,16},{33,31}};
    for (int i = 0; i < 6; ++i) {
        int n = stages[i][0], cnt = stages[i][1];
        stu_q  <<<(D_ / 16) * (D_ / 16) / 8, 256, 0, stream>>>(Psi, myt, Q, n);
        stu_psi<<<dim3((D_ / 16) * (D_ / 16) / 8, cnt), 256, 0, stream>>>(Psi, Q, n);
    }

    // spectral path
    stu_bt  <<<KF_ * (L_ / 64) * (D_ / 16) / 8, 256, 0, stream>>>(ubf, mphiT, Bt);
    stu_dphi<<<(L_ / 64) * (D_ / 16) / 8, 256, 0, stream>>>(Af, Bt, dphi);
    stu_ar  <<<(L_ / 64) * (D_ / 16) / 8, 256, 0, stream>>>(ubf, mut, dphi, dbf);

    // scan as windowed GEMM
    stu_final<<<(L_ / 64) * (D_ / 16) / 8, 256, 0, stream>>>(dbf, Psi, y);
}
